// DCNModule_40458591928628
// MI455X (gfx1250) — matrix operand order repaired and checked by reading, compiled, not run on hardware
//
#include <hip/hip_runtime.h>
#include <math.h>

// ---------------------------------------------------------------------------
// Problem constants (from reference): B=8, C=256, H=W=64, O=256, 3x3, pad 1
// ---------------------------------------------------------------------------
#define BB 8
#define CC 256
#define HH 64
#define WW 64
#define OO 256
#define PP 4096            // H*W
#define KT 9               // 3x3 taps
#define NSTEPS 72          // CK / 32 = 2304/32
#define GNG 32
#define GSIZE 32768        // (O/GNG)*P = 8*4096

typedef __attribute__((ext_vector_type(16))) __bf16 v16bf;
typedef __attribute__((ext_vector_type(8)))  float  v8f;

union FragBF { uint4 u[2]; v16bf v; };

__device__ __forceinline__ unsigned short f2bf(float f) {
    unsigned u = __builtin_bit_cast(unsigned, f);
    unsigned r = u + 0x7FFFu + ((u >> 16) & 1u);   // round-to-nearest-even
    return (unsigned short)(r >> 16);
}

// ---------------------------------------------------------------------------
// Kernel 1: offset conv (18ch 3x3) -> sampling coords (py,px) per (b,kk,p)
// ---------------------------------------------------------------------------
__global__ __launch_bounds__(256) void dcn_offsets(
    const float* __restrict__ x, const float* __restrict__ w_off,
    const float* __restrict__ b_off, float2* __restrict__ pyx)
{
    int tid = blockIdx.x * 256 + threadIdx.x;   // 32768 = B*P
    int b = tid >> 12, p = tid & 4095;
    int ho = p >> 6, wo = p & 63;

    float acc[18];
    #pragma unroll
    for (int ch = 0; ch < 18; ++ch) acc[ch] = b_off[ch];

    const float* xb = x + ((size_t)b << 20);     // b * C*P
    for (int c = 0; c < CC; ++c) {
        float x9[9];
        const float* pl = xb + (c << 12);
        #pragma unroll
        for (int kh = 0; kh < 3; ++kh)
            #pragma unroll
            for (int kw = 0; kw < 3; ++kw) {
                int y = ho - 1 + kh, xx = wo - 1 + kw;
                bool ok = (y >= 0) & (y < HH) & (xx >= 0) & (xx < WW);
                x9[kh * 3 + kw] = ok ? pl[y * WW + xx] : 0.f;
            }
        #pragma unroll
        for (int ch = 0; ch < 18; ++ch) {
            const float* w9 = w_off + (((ch << 8) + c) * 9);
            float a = acc[ch];
            #pragma unroll
            for (int t = 0; t < 9; ++t) a = fmaf(x9[t], w9[t], a);
            acc[ch] = a;
        }
    }
    #pragma unroll
    for (int kk = 0; kk < KT; ++kk) {
        float py = acc[2 * kk]     + (float)(ho - 1 + kk / 3);
        float px = acc[2 * kk + 1] + (float)(wo - 1 + kk % 3);
        pyx[((b * KT + kk) << 12) + p] = make_float2(py, px);
    }
}

// ---------------------------------------------------------------------------
// Kernel 2: repack weights to bf16, K-step-major / M-major layout:
//   wG[ks][o][klocal],  kglobal = kk*256 + c = ks*32 + klocal
// ---------------------------------------------------------------------------
__global__ __launch_bounds__(256) void dcn_wcvt(
    const float* __restrict__ w_dcn, unsigned short* __restrict__ wG)
{
    int kg = blockIdx.x;          // 0..2303
    int o  = threadIdx.x;         // 0..255
    int ks = kg >> 5, kl = kg & 31;
    int kk = kg >> 8, c  = kg & 255;
    wG[(ks << 13) + (o << 5) + kl] = f2bf(w_dcn[((o << 8) + c) * 9 + kk]);
}

// ---------------------------------------------------------------------------
// Kernel 3: implicit deformable GEMM with WMMA (bf16 -> f32)
//   block tile: M=256 (all O) x N=32 positions; 8 waves, each 32x32 accum
//   A tile staged with CDNA5 async global->LDS copies (ASYNCcnt-tracked)
// ---------------------------------------------------------------------------
#define ASTR 40   // padded LDS row stride (halfwords) for bank spread
#define BSTR 40

__global__ __launch_bounds__(256) void dcn_gemm(
    const float* __restrict__ x, const unsigned short* __restrict__ wG,
    const float2* __restrict__ pyx, float* __restrict__ out)
{
    __shared__ unsigned short Alds[256 * ASTR];
    __shared__ unsigned short Blds[32 * BSTR];
    __shared__ int   sOff[32][4];
    __shared__ float sWgt[32][4];

    const int tid  = threadIdx.x;
    const int lane = tid & 31;
    const int wv   = tid >> 5;       // 8 waves
    const int m0   = wv << 5;        // wave owns M rows m0..m0+31
    const int b    = blockIdx.x >> 7;
    const int p0   = (blockIdx.x & 127) << 5;

    const float* xb = x + ((size_t)b << 20);

    v8f acc[2][2] = {};

    const int hi   = lane >> 4;
    const int lrow = lane & 15;

    // per-thread LDS byte offset of this thread's A row (low 32 bits of the
    // generic pointer ARE the LDS address per ISA 10.2)
    const unsigned ldsA = (unsigned)(size_t)(&Alds[tid * ASTR]);

    for (int ks = 0; ks < NSTEPS; ++ks) {
        const int kk = ks >> 3;
        const int cs = ks & 7;
        const int c0 = cs << 5;

        // ---- stage A tile: async global->LDS, 2x16B per lane (one row) ----
        {
            unsigned long long ga =
                (unsigned long long)(const void*)(wG + (ks << 13) + (tid << 5));
            asm volatile(
                "global_load_async_to_lds_b128 %0, %1, off\n\t"
                "global_load_async_to_lds_b128 %0, %1, off offset:16"
                :: "v"(ldsA), "v"(ga) : "memory");
        }
        if (ks + 1 < NSTEPS)
            __builtin_prefetch(wG + ((ks + 1) << 13) + (tid << 5), 0, 3);

        // ---- recompute bilinear metadata when tap changes ----
        if (cs == 0 && tid < 32) {
            float2 v = pyx[((b * KT + kk) << 12) + p0 + tid];
            float py = v.x, px = v.y;
            float fy = floorf(py), fx = floorf(px);
            int y0 = (int)fy, x0 = (int)fx;
            float dy = py - fy, dx = px - fx;
            float w4[4] = { (1.f - dy) * (1.f - dx), (1.f - dy) * dx,
                            dy * (1.f - dx),         dy * dx };
            int yy[4] = { y0, y0, y0 + 1, y0 + 1 };
            int xx[4] = { x0, x0 + 1, x0, x0 + 1 };
            #pragma unroll
            for (int j = 0; j < 4; ++j) {
                bool ok = (yy[j] >= 0) & (yy[j] < HH) & (xx[j] >= 0) & (xx[j] < WW);
                int yc = min(max(yy[j], 0), HH - 1);
                int xc = min(max(xx[j], 0), WW - 1);
                sOff[tid][j] = yc * WW + xc;
                sWgt[tid][j] = ok ? w4[j] : 0.f;
            }
        }
        __syncthreads();   // meta visible (async A copy still in flight)

        // ---- build B tile: 32 n x 32 c bilinear samples (4 per thread) ----
        // overlaps with the in-flight async A-tile copy
        {
            const int n     = tid >> 3;
            const int cbase = (tid & 7) << 2;
            const int o00 = sOff[n][0], o01 = sOff[n][1];
            const int o10 = sOff[n][2], o11 = sOff[n][3];
            const float w00 = sWgt[n][0], w01 = sWgt[n][1];
            const float w10 = sWgt[n][2], w11 = sWgt[n][3];
            unsigned short vals[4];
            #pragma unroll
            for (int j = 0; j < 4; ++j) {
                const float* pl = xb + ((c0 + cbase + j) << 12);
                float v = w00 * pl[o00] + w01 * pl[o01]
                        + w10 * pl[o10] + w11 * pl[o11];
                vals[j] = f2bf(v);
            }
            uint2 pk;
            pk.x = (unsigned)vals[0] | ((unsigned)vals[1] << 16);
            pk.y = (unsigned)vals[2] | ((unsigned)vals[3] << 16);
            *(uint2*)(&Blds[n * BSTR + cbase]) = pk;
        }
        // own async A-row has landed in LDS; barrier makes all rows visible
        asm volatile("s_wait_asynccnt 0x0" ::: "memory");
        __syncthreads();

        // ---- fragment loads (match ISA 16-bit A/B VGPR layouts) + WMMA ----
        FragBF afr[2], bfr[2];
        #pragma unroll
        for (int mi = 0; mi < 2; ++mi) {
            int m = m0 + mi * 16 + lrow;
            const uint4* ap = (const uint4*)(&Alds[m * ASTR + hi * 8]);
            afr[mi].u[0] = ap[0];        // K = base .. base+7   (VGPR0-3)
            afr[mi].u[1] = ap[2];        // K = base+16..base+23 (VGPR4-7)
        }
        #pragma unroll
        for (int ni = 0; ni < 2; ++ni) {
            int n = ni * 16 + lrow;
            const uint4* bp = (const uint4*)(&Blds[n * BSTR + hi * 8]);
            bfr[ni].u[0] = bp[0];        // K = hi*8 .. +7
            bfr[ni].u[1] = bp[2];        // K = 16+hi*8 .. +7
        }
        #pragma unroll
        for (int mi = 0; mi < 2; ++mi)
            #pragma unroll
            for (int ni = 0; ni < 2; ++ni)
                acc[mi][ni] = __builtin_amdgcn_wmma_f32_16x16x32_bf16(
                    false, afr[mi].v, false, bfr[ni].v,
                    (short)0, acc[mi][ni], false, false);
        __syncthreads();
    }

    // ---- epilogue: C/D layout -> out[b][M][p] ----
    #pragma unroll
    for (int mi = 0; mi < 2; ++mi)
        #pragma unroll
        for (int ni = 0; ni < 2; ++ni) {
            int mbase = m0 + mi * 16 + hi * 8;
            float* op = out + (((size_t)b * OO + mbase) << 12) + p0 + ni * 16 + lrow;
            #pragma unroll
            for (int r = 0; r < 8; ++r)
                op[(size_t)r << 12] = acc[mi][ni][r];
        }
}

// ---------------------------------------------------------------------------
// Kernel 4: fused GroupNorm + Mish, one block per (b, group)
// ---------------------------------------------------------------------------
__global__ __launch_bounds__(256) void dcn_gn_mish(
    float* __restrict__ y, const float* __restrict__ gamma,
    const float* __restrict__ beta)
{
    int bg = blockIdx.x;               // 0..255
    int b = bg >> 5, g = bg & 31;
    float* base = y + (((size_t)b * OO + g * 8) << 12);

    float s = 0.f, q = 0.f;
    for (int i = threadIdx.x; i < GSIZE; i += 256) {
        float v = base[i];
        s += v; q += v * v;
    }
    __shared__ float ss[256], sq[256];
    ss[threadIdx.x] = s; sq[threadIdx.x] = q;
    __syncthreads();
    for (int st = 128; st > 0; st >>= 1) {
        if (threadIdx.x < st) {
            ss[threadIdx.x] += ss[threadIdx.x + st];
            sq[threadIdx.x] += sq[threadIdx.x + st];
        }
        __syncthreads();
    }
    __shared__ float sMean, sRstd;
    if (threadIdx.x == 0) {
        float m = ss[0] * (1.f / GSIZE);
        float var = sq[0] * (1.f / GSIZE) - m * m;
        sMean = m;
        sRstd = rsqrtf(var + 1e-5f);
    }
    __syncthreads();
    float m = sMean, rs = sRstd;
    for (int i = threadIdx.x; i < GSIZE; i += 256) {
        int o = g * 8 + (i >> 12);
        float v = (base[i] - m) * rs * gamma[o] + beta[o];
        float sp = (v > 20.f) ? v : log1pf(expf(v));
        base[i] = v * tanhf(sp);
    }
}

// ---------------------------------------------------------------------------
// Host orchestration
// ---------------------------------------------------------------------------
extern "C" void kernel_launch(void* const* d_in, const int* in_sizes, int n_in,
                              void* d_out, int out_size, void* d_ws, size_t ws_size,
                              hipStream_t stream)
{
    const float* x     = (const float*)d_in[0];
    const float* w_off = (const float*)d_in[1];
    const float* b_off = (const float*)d_in[2];
    const float* w_dcn = (const float*)d_in[3];
    const float* gamma = (const float*)d_in[4];
    const float* beta  = (const float*)d_in[5];
    float* out = (float*)d_out;

    // workspace: [pyx float2: B*KT*P] [wG bf16: 2304*256]
    float2* pyx = (float2*)d_ws;
    unsigned short* wG =
        (unsigned short*)((char*)d_ws + (size_t)BB * KT * PP * sizeof(float2));

    dcn_offsets<<<(BB * PP) / 256, 256, 0, stream>>>(x, w_off, b_off, pyx);
    dcn_wcvt<<<2304, 256, 0, stream>>>(w_dcn, wG);
    dcn_gemm<<<BB * (PP / 32), 256, 0, stream>>>(x, wG, pyx, out);
    dcn_gn_mish<<<BB * GNG, 256, 0, stream>>>(out, gamma, beta);
}
